// VectorQuantizer_noLinear_1271310319903
// MI455X (gfx1250) — matrix operand order repaired and checked by reading, not compiled, not run on hardware
//
#include <hip/hip_runtime.h>

// ---------------------------------------------------------------------------
// VectorQuantizer (rotation trick) for MI455X / gfx1250.
// Dominant cost: 32768x8192x768 distance GEMM -> bf16 WMMA (f32 accum).
// znorm is dropped from the argmin (constant per row). Distance matrix is
// never materialized: running argmin kept in registers.
// Wave tile = 32x64 (2 M-frags x 4 N-frags): 1 ds_load_b128 per WMMA.
// ---------------------------------------------------------------------------

typedef __attribute__((ext_vector_type(16))) __bf16        v16bf;
typedef __attribute__((ext_vector_type(8)))  float         v8f;
typedef __attribute__((ext_vector_type(8)))  unsigned int  v8u;

#define ND      768
#define NROWS   32768       // 32*1024
#define NCODES  8192
#define KT      24          // 768 / 32 k-steps
#define OUT_ELEMS 25165824  // 32*1024*768

// ---- float -> bf16 pair (round to nearest even) ---------------------------
__device__ __forceinline__ unsigned int bf16pair(float lo, float hi) {
  unsigned int bl = __float_as_uint(lo);
  unsigned int bh = __float_as_uint(hi);
  bl = (bl + 0x7FFFu + ((bl >> 16) & 1u)) >> 16;
  bh = (bh + 0x7FFFu + ((bh >> 16) & 1u)) >> 16;
  return bl | (bh << 16);
}

// ---------------------------------------------------------------------------
// Pack z (f32 row-major) into WMMA A-fragment layout (16x32 bf16 per frag).
// Frag id = mf*24 + kt. Per ISA 7.12.2 (16-bit A 16x32):
//   lane l: row = 16*mf + (l&15), half = l>>4
//   slot v: k = kt*32 + 8*half + (v<4 ? 2v : 16+2(v-4)), pair (k,k+1)
// Stored as [frag][lane][v] -> each lane reads 32 contiguous bytes.
// ---------------------------------------------------------------------------
__global__ void pack_z_kernel(const float* __restrict__ z,
                              unsigned int* __restrict__ zpk) {
  int t    = blockIdx.x * blockDim.x + threadIdx.x;   // one thread per (frag,lane)
  int lane = t & 31;
  int frag = t >> 5;                                  // mf*24 + kt
  int kt   = frag % KT;
  int mf   = frag / KT;
  int row  = (mf << 4) + (lane & 15);
  int half = lane >> 4;
  const float* src = z + row * ND + kt * 32 + half * 8;
  unsigned int* dst = zpk + frag * 256 + lane * 8;
#pragma unroll
  for (int v = 0; v < 8; ++v) {
    int kk = (v < 4) ? (2 * v) : (16 + 2 * (v - 4));
    dst[v] = bf16pair(src[kk], src[kk + 1]);
  }
}

// ---------------------------------------------------------------------------
// Pack embedding into WMMA B-fragment layout (32x16 bf16 per frag, B = emb^T):
//   lane l: col n, khalf = l>>4; slot v holds k = kt*32 + 8*khalf + (v<4 ? 2v : 16+2(v-4)), pair.
// Frag order: frag = ((nt2*24 + kt)*8 + fi) so the 8 fragments of one
// (128-code, 32-k) tile are 8KB contiguous -> trivial cooperative LDS copy.
// ---------------------------------------------------------------------------
__global__ void pack_emb_kernel(const float* __restrict__ emb,
                                unsigned int* __restrict__ epk) {
  int t    = blockIdx.x * blockDim.x + threadIdx.x;
  int lane = t & 31;
  int frag = t >> 5;                 // ((nt2*24 + kt)*8 + fi)
  int fi   = frag & 7;
  int tmp  = frag >> 3;              // nt2*24 + kt
  int kt   = tmp % KT;
  int nt2  = tmp / KT;
  int n    = nt2 * 128 + fi * 16 + (lane & 15);
  int khalf = lane >> 4;
  const float* src = emb + n * ND + kt * 32 + khalf * 8;
  unsigned int* dst = epk + frag * 256 + lane * 8;
#pragma unroll
  for (int v = 0; v < 8; ++v) {
    int kk = (v < 4) ? (2 * v) : (16 + 2 * (v - 4));
    dst[v] = bf16pair(src[kk], src[kk + 1]);
  }
}

// ---- exact f32 code norms (argmin offsets) --------------------------------
__global__ void emb_norm_kernel(const float* __restrict__ emb,
                                float* __restrict__ enorm) {
  int r = blockIdx.x * blockDim.x + threadIdx.x;
  if (r >= NCODES) return;
  const float4* p = (const float4*)(emb + r * ND);
  float s = 0.0f;
#pragma unroll 4
  for (int i = 0; i < ND / 4; ++i) {
    float4 v = p[i];
    s += v.x * v.x + v.y * v.y + v.z * v.z + v.w * v.w;
  }
  enorm[r] = s;
}

// ---------------------------------------------------------------------------
// GEMM + running argmin. 256 blocks x 8 waves; block tile = 128 rows.
// Wave (pw, nh): pw = wave>>1 picks an M-pair (32 rows, 2 A-frags),
// nh = wave&1 picks the 64-code half of the 128-code N-tile (4 B-frags).
// Per k-step: 8KB B tile staged in LDS (shared 4-ways per fragment),
// 2 A fragments per wave from global (hits L2), 8 WMMAs with only
// 8 ds_load_b128 -> 1 DS load per v_wmma_f32_16x16x32_bf16.
// ---------------------------------------------------------------------------
__launch_bounds__(256, 1)
__global__ void vq_argmin_kernel(const unsigned int* __restrict__ zpk,
                                 const unsigned int* __restrict__ epk,
                                 const float* __restrict__ enorm,
                                 int* __restrict__ indices) {
  __shared__ __align__(32) unsigned int sB[2048];   // 8 KB: 8 B-fragments
  const int lane = threadIdx.x & 31;
  const int wave = threadIdx.x >> 5;
  const int pw   = wave >> 1;                 // M-pair 0..3
  const int nh   = wave & 1;                  // N-half 0..1
  const int mf0  = blockIdx.x * 8 + pw * 2;   // wave's two M-fragments
  const unsigned int* aBase0 = zpk + (size_t)mf0 * KT * 256 + lane * 8;
  const unsigned int* aBase1 = aBase0 + (size_t)KT * 256;
  const int nl = lane & 15;

  float bestD[2][8];
  int   bestI[2][8];
#pragma unroll
  for (int m = 0; m < 2; ++m)
#pragma unroll
    for (int v = 0; v < 8; ++v) { bestD[m][v] = 3.4e38f; bestI[m][v] = 0; }

  for (int nt2 = 0; nt2 < NCODES / 128; ++nt2) {
    v8f acc[2][4];
#pragma unroll
    for (int m = 0; m < 2; ++m)
#pragma unroll
      for (int a = 0; a < 4; ++a)
#pragma unroll
        for (int v = 0; v < 8; ++v) acc[m][a][v] = 0.0f;

    for (int kt = 0; kt < KT; ++kt) {
      __syncthreads();                       // previous tile fully consumed
      const uint4* gsrc = (const uint4*)(epk + (size_t)(nt2 * KT + kt) * 2048);
      uint4* ldst = (uint4*)sB;
      ldst[threadIdx.x]       = gsrc[threadIdx.x];
      ldst[threadIdx.x + 256] = gsrc[threadIdx.x + 256];
      __syncthreads();

      v16bf a0 = __builtin_bit_cast(v16bf, *(const v8u*)(aBase0 + kt * 256));
      v16bf a1 = __builtin_bit_cast(v16bf, *(const v8u*)(aBase1 + kt * 256));
#pragma unroll
      for (int a = 0; a < 4; ++a) {
        v8u   braw  = *(const v8u*)(sB + (nh * 4 + a) * 256 + lane * 8);
        v16bf bfrag = __builtin_bit_cast(v16bf, braw);
        acc[0][a] = __builtin_amdgcn_wmma_f32_16x16x32_bf16(
            false, a0, false, bfrag, (short)0, acc[0][a], false, false);
        acc[1][a] = __builtin_amdgcn_wmma_f32_16x16x32_bf16(
            false, a1, false, bfrag, (short)0, acc[1][a], false, false);
      }
    }

    // dist(m,n) - ||z_m||^2 = ||e_n||^2 - 2*dot ; ascending n => strict '<'
    // keeps the first-minimum, matching jnp.argmin semantics.
#pragma unroll
    for (int a = 0; a < 4; ++a) {
      int n = nt2 * 128 + (nh * 4 + a) * 16 + nl;
      float en = enorm[n];
#pragma unroll
      for (int m = 0; m < 2; ++m)
#pragma unroll
        for (int v = 0; v < 8; ++v) {
          float d = en - 2.0f * acc[m][a][v];
          if (d < bestD[m][v]) { bestD[m][v] = d; bestI[m][v] = n; }
        }
    }
  }

  // Cross-lane min within each 16-lane half (lanes of one half share rows;
  // D layout: lane l, vgpr v -> m = v + 8*(l>>4), n = l&15). Tie -> lower n.
  // The two 64-code halves of each N-tile live in different waves, so the
  // final combine goes through LDS (disjoint n => '<' + lower-idx tie-break).
  __shared__ float sD[2][128];   // [mplane rows via pw] packed per wave-pair
  __shared__ int   sI[2][128];
#pragma unroll
  for (int m = 0; m < 2; ++m)
#pragma unroll
    for (int v = 0; v < 8; ++v) {
      float d = bestD[m][v];
      int   i = bestI[m][v];
#pragma unroll
      for (int off = 1; off < 16; off <<= 1) {
        float od = __shfl_xor(d, off, 32);
        int   oi = __shfl_xor(i, off, 32);
        if (od < d || (od == d && oi < i)) { d = od; i = oi; }
      }
      bestD[m][v] = d; bestI[m][v] = i;
    }
  __syncthreads();
  if (nl == 0 && nh == 0) {       // N-half 0 publishes its candidates
#pragma unroll
    for (int m = 0; m < 2; ++m)
#pragma unroll
      for (int v = 0; v < 8; ++v) {
        int slot = (pw * 2 + m) * 16 + 8 * (lane >> 4) + v;   // row within block
        sD[0][slot] = bestD[m][v];
        sI[0][slot] = bestI[m][v];
      }
  }
  __syncthreads();
  if (nl == 0 && nh == 1) {       // N-half 1 combines and writes result
#pragma unroll
    for (int m = 0; m < 2; ++m)
#pragma unroll
      for (int v = 0; v < 8; ++v) {
        int slot = (pw * 2 + m) * 16 + 8 * (lane >> 4) + v;
        float d0 = sD[0][slot];
        int   i0 = sI[0][slot];
        float d1 = bestD[m][v];
        int   i1 = bestI[m][v];
        bool take0 = (d0 < d1) || (d0 == d1 && i0 < i1);
        indices[blockIdx.x * 128 + slot] = take0 ? i0 : i1;
      }
  }
}

// ---------------------------------------------------------------------------
// Rotation trick + loss. One block (256 thr) per row; 3 elems/thread.
// All rotation scalars derive from {Σe², Σq², Σe·q}.
// ---------------------------------------------------------------------------
__launch_bounds__(256, 1)
__global__ void rotate_out_kernel(const float* __restrict__ z,
                                  const float* __restrict__ emb,
                                  const int* __restrict__ indices,
                                  float* __restrict__ out,
                                  float* __restrict__ lossAcc) {
  const int row = blockIdx.x;
  const int tid = threadIdx.x;
  const float* e = z + (size_t)row * ND;
  const float* q = emb + (size_t)indices[row] * ND;

  float ev[3], qv[3];
  float s_ee = 0.0f, s_qq = 0.0f, s_eq = 0.0f;
#pragma unroll
  for (int j = 0; j < 3; ++j) {
    int i = tid + j * 256;
    float a = e[i], b = q[i];
    ev[j] = a; qv[j] = b;
    s_ee += a * a; s_qq += b * b; s_eq += a * b;
  }
#pragma unroll
  for (int off = 16; off; off >>= 1) {
    s_ee += __shfl_xor(s_ee, off, 32);
    s_qq += __shfl_xor(s_qq, off, 32);
    s_eq += __shfl_xor(s_eq, off, 32);
  }
  __shared__ float red[3][8];
  int wave = tid >> 5, lane = tid & 31;
  if (lane == 0) { red[0][wave] = s_ee; red[1][wave] = s_qq; red[2][wave] = s_eq; }
  __syncthreads();
  if (tid == 0) {
    float a = 0.0f, b = 0.0f, c = 0.0f;
#pragma unroll
    for (int w = 0; w < 8; ++w) { a += red[0][w]; b += red[1][w]; c += red[2][w]; }
    red[0][0] = a; red[1][0] = b; red[2][0] = c;
  }
  __syncthreads();
  float SEE = red[0][0], SQQ = red[1][0], SEQ = red[2][0];

  float ns = sqrtf(SEE), ntg = sqrtf(SQQ);
  float inv_ns = 1.0f / fmaxf(ns, 1e-6f);
  float inv_nt = 1.0f / fmaxf(ntg, 1e-6f);
  float nw2 = SEE * inv_ns * inv_ns + SQQ * inv_nt * inv_nt
            + 2.0f * SEQ * inv_ns * inv_nt;            // ||u+q||^2
  float nw = sqrtf(fmaxf(nw2, 0.0f));
  float inv_nw = 1.0f / fmaxf(nw, 1e-6f);
  float ew = (SEE * inv_ns + SEQ * inv_nt) * inv_nw;   // e . w
  float eu = SEE * inv_ns;                             // e . u
  float scale = ntg * inv_ns;                          // ||q|| / max(||e||,eps)
  float cw = 2.0f * ew * inv_nw;                       // coeff of (u_i + q_i)

#pragma unroll
  for (int j = 0; j < 3; ++j) {
    int i = tid + j * 256;
    float ui = ev[j] * inv_ns;
    float qi = qv[j] * inv_nt;
    float r  = ev[j] - cw * (ui + qi) + 2.0f * eu * qi;
    out[(size_t)row * ND + i] = r * scale;
  }
  if (tid == 0) {
    float sdiff = SQQ - 2.0f * SEQ + SEE;              // Σ (q - e)^2 this row
    atomicAdd(lossAcc, sdiff * (2.0f / (float)OUT_ELEMS));
  }
}

__global__ void zero_loss_kernel(float* p) {
  if (blockIdx.x == 0 && threadIdx.x == 0) *p = 0.0f;
}

// ---------------------------------------------------------------------------
extern "C" void kernel_launch(void* const* d_in, const int* in_sizes, int n_in,
                              void* d_out, int out_size, void* d_ws, size_t ws_size,
                              hipStream_t stream) {
  const float* z   = (const float*)d_in[0];   // [32,1024,768] f32
  const float* emb = (const float*)d_in[1];   // [8192,768]    f32
  float* out = (float*)d_out;

  // Workspace layout (~63.1 MB total)
  char* ws = (char*)d_ws;
  unsigned int* zpk = (unsigned int*)ws;                              // 50,331,648 B
  unsigned int* epk = (unsigned int*)(ws + 50331648);                 // 12,582,912 B
  float* enorm      = (float*)(ws + 50331648 + 12582912);             //     32,768 B
  int*   indices    = (int*)  (ws + 50331648 + 12582912 + 32768);     //    131,072 B
  float* lossAcc    = out + (out_size - 1);

  zero_loss_kernel<<<1, 32, 0, stream>>>(lossAcc);
  // 2048 m-frags * 24 kt * 32 lanes = 1,572,864 threads
  pack_z_kernel<<<6144, 256, 0, stream>>>(z, zpk);
  // 512 n-frags * 24 kt * 32 lanes = 393,216 threads
  pack_emb_kernel<<<1536, 256, 0, stream>>>(emb, epk);
  emb_norm_kernel<<<NCODES / 256, 256, 0, stream>>>(emb, enorm);
  vq_argmin_kernel<<<NROWS / 128, 256, 0, stream>>>(zpk, epk, enorm, indices);
  rotate_out_kernel<<<NROWS, 256, 0, stream>>>(z, emb, indices, out, lossAcc);
}
